// SelfAttention_18846316495125
// MI455X (gfx1250) — matrix operand order repaired and checked by reading, not compiled, not run on hardware
//
#include <hip/hip_runtime.h>
#include <hip/hip_bf16.h>

// Problem constants (reference: B=4, S=2048, E=1024)
#define NB 4
#define NS 2048
#define NE 1024
#define LDT 72  // LDS tile row stride in elements (144B = 9 x 16B chunks)

typedef __bf16 bf16_t;
typedef bf16_t v16bf __attribute__((ext_vector_type(16)));
typedef bf16_t v8bf  __attribute__((ext_vector_type(8)));
typedef float  v8f   __attribute__((ext_vector_type(8)));

static __device__ __forceinline__ v8f wmma_bf16(v16bf a, v16bf b, v8f c) {
  // (neg_a, A, neg_b, B, c_mod, C, reuse_a, reuse_b)
  return __builtin_amdgcn_wmma_f32_16x16x32_bf16(false, a, false, b, (short)0, c,
                                                 false, false);
}

static __device__ __forceinline__ v16bf cat8(v8bf lo, v8bf hi) {
  return __builtin_shufflevector(lo, hi, 0, 1, 2, 3, 4, 5, 6, 7, 8, 9, 10, 11,
                                 12, 13, 14, 15);
}

// A fragment (16x32 bf16, ISA 16-bit A layout), LDS row stride = LDT elems.
// lane<16 : row = lane,    K in {0..7} U {16..23}
// lane>=16: row = lane-16, K in {8..15} U {24..31}
static __device__ __forceinline__ v16bf load_a_frag(const bf16_t* p, int lane) {
  const int r  = lane & 15;
  const int kb = (lane < 16) ? 0 : 8;
  v8bf lo = *(const v8bf*)(p + r * LDT + kb);
  v8bf hi = *(const v8bf*)(p + r * LDT + kb + 16);
  return cat8(lo, hi);
}

// B fragment (32x16 bf16), LDS stored [n][k], row stride = LDT elems.
// lane<16 : col = lane, K in {0..7} U {16..23} ; lane>=16: col = lane-16, K in {8..15} U {24..31}
static __device__ __forceinline__ v16bf load_b_frag(const bf16_t* p, int lane) {
  const int n  = lane & 15;
  const int kb = (lane < 16) ? 0 : 8;
  v8bf lo = *(const v8bf*)(p + n * LDT + kb);
  v8bf hi = *(const v8bf*)(p + n * LDT + kb + 16);
  return cat8(lo, hi);
}

// ---------------------------------------------------------------------------
// Shared GEMM core. Both operands are K-contiguous ("NT" style):
//   Ab: rows gm0..gm0+127 (row-major, leading dim ldA)
//   Bb: rows gn0..gn0+127 (row-major, leading dim ldB; each row = out column)
// Block tile 128x128, BK=64, 128 threads = 4 waves in a 2x2 grid.
// Wave tile 64x64 -> acc[4][4] 16x16 f32 fragments (128 accum VGPRs).
// Per K-step: 4 A-frags + 4 B-frags (16 ds_load_b128) feed 16 WMMAs.
// ---------------------------------------------------------------------------
static __device__ __forceinline__ void gemm_core(
    const bf16_t* __restrict__ Ab, int ldA, const bf16_t* __restrict__ Bb,
    int ldB, int K, int gm0, int gn0, int tid, int lane, int wm, int wn,
    v8f acc[4][4]) {
  __shared__ __attribute__((aligned(16))) bf16_t lA[128 * LDT];
  __shared__ __attribute__((aligned(16))) bf16_t lB[128 * LDT];

  for (int k0 = 0; k0 < K; k0 += 64) {
    // Stage both 128x64 tiles: 1024 16B-chunks each, 8 per thread per tile.
#pragma unroll
    for (int i = 0; i < 4; ++i) {
      const int idx = tid + i * 128;
      const int row = idx >> 2;
      const int kc  = (idx & 3) * 16;
      *(v8bf*)(lA + row * LDT + kc) =
          *(const v8bf*)(Ab + (size_t)(gm0 + row) * ldA + k0 + kc);
      *(v8bf*)(lA + row * LDT + kc + 8) =
          *(const v8bf*)(Ab + (size_t)(gm0 + row) * ldA + k0 + kc + 8);
      *(v8bf*)(lB + row * LDT + kc) =
          *(const v8bf*)(Bb + (size_t)(gn0 + row) * ldB + k0 + kc);
      *(v8bf*)(lB + row * LDT + kc + 8) =
          *(const v8bf*)(Bb + (size_t)(gn0 + row) * ldB + k0 + kc + 8);
    }
    // Prefetch next K-tile (lowers to global_prefetch_b8).
    if (k0 + 64 < K) {
      __builtin_prefetch(Ab + (size_t)(gm0 + tid) * ldA + k0 + 64, 0, 1);
      __builtin_prefetch(Bb + (size_t)(gn0 + tid) * ldB + k0 + 64, 0, 1);
    }
    __syncthreads();
#pragma unroll
    for (int ks = 0; ks < 2; ++ks) {
      v16bf bfr[4];
#pragma unroll
      for (int nf = 0; nf < 4; ++nf)
        bfr[nf] = load_b_frag(lB + (wn + nf * 16) * LDT + ks * 32, lane);
#pragma unroll
      for (int mf = 0; mf < 4; ++mf) {
        v16bf af = load_a_frag(lA + (wm + mf * 16) * LDT + ks * 32, lane);
#pragma unroll
        for (int nf = 0; nf < 4; ++nf)
          acc[mf][nf] = wmma_bf16(af, bfr[nf], acc[mf][nf]);
      }
    }
    __syncthreads();
  }
}

// ---------------------------------------------------------------------------
// fp32 -> bf16 convert, 8 elements per thread (vector loads/stores)
// ---------------------------------------------------------------------------
__global__ __launch_bounds__(256) void cvt8_kernel(
    const float* __restrict__ src, bf16_t* __restrict__ dst, int n8) {
  const int i = blockIdx.x * 256 + threadIdx.x;
  if (i < n8) {
    const float4* s = (const float4*)src + (size_t)i * 2;
    const float4 a  = s[0];
    const float4 b  = s[1];
    v8bf o = {(bf16_t)a.x, (bf16_t)a.y, (bf16_t)a.z, (bf16_t)a.w,
              (bf16_t)b.x, (bf16_t)b.y, (bf16_t)b.z, (bf16_t)b.w};
    ((v8bf*)dst)[i] = o;
  }
}

// ---------------------------------------------------------------------------
// Fused convert + transpose for weights: fp32 W[K][N] -> bf16 Wt[N][K]
// ---------------------------------------------------------------------------
__global__ __launch_bounds__(256) void wtcvt_kernel(
    const float* __restrict__ W, bf16_t* __restrict__ Wt, int Kd, int Nd) {
  __shared__ bf16_t t[32][33];
  const int tx = threadIdx.x & 31;
  const int ty = threadIdx.x >> 5;
  const int k0 = blockIdx.y * 32;
  const int n0 = blockIdx.x * 32;
#pragma unroll
  for (int i = 0; i < 32; i += 8)
    t[ty + i][tx] = (bf16_t)W[(size_t)(k0 + ty + i) * Nd + n0 + tx];
  __syncthreads();
#pragma unroll
  for (int i = 0; i < 32; i += 8)
    Wt[(size_t)(n0 + ty + i) * Kd + k0 + tx] = t[tx][ty + i];
}

// ---------------------------------------------------------------------------
// Projection GEMM: C[M=B*S, N=E] = X @ Wt^T + bias  (Wt is [N][K])
// TRANS_OUT=false: C row-major bf16 [M,N] (Q, K projections)
// TRANS_OUT=true : C written as Vt[b][e][s] bf16 (V projection)
// ---------------------------------------------------------------------------
template <bool TRANS_OUT>
__global__ __launch_bounds__(128) void proj_kernel(
    const bf16_t* __restrict__ A, const bf16_t* __restrict__ Wt,
    const float* __restrict__ bias, bf16_t* __restrict__ C) {
  const int tid  = threadIdx.x;
  const int lane = tid & 31;
  const int wid  = tid >> 5;
  const int wm   = (wid >> 1) * 64;
  const int wn   = (wid & 1) * 64;
  const int gm0  = blockIdx.x * 128;
  const int gn0  = blockIdx.y * 128;

  v8f acc[4][4] = {};
  gemm_core(A, NE, Wt, NE, NE, gm0, gn0, tid, lane, wm, wn, acc);

  const int col   = lane & 15;
  const int rbase = (lane < 16) ? 0 : 8;
#pragma unroll
  for (int mf = 0; mf < 4; ++mf)
#pragma unroll
    for (int nf = 0; nf < 4; ++nf) {
      const int gc   = gn0 + wn + nf * 16 + col;
      const float bv = bias[gc];
#pragma unroll
      for (int r = 0; r < 8; ++r) {
        const int gr      = gm0 + wm + mf * 16 + rbase + r;
        const bf16_t outv = (bf16_t)(acc[mf][nf][r] + bv);
        if (TRANS_OUT) {
          const int b = gr >> 11;       // gr / NS
          const int s = gr & (NS - 1);  // gr % NS
          C[((size_t)b * NE + gc) * NS + s] = outv;
        } else {
          C[(size_t)gr * NE + gc] = outv;
        }
      }
    }
}

// ---------------------------------------------------------------------------
// Scores GEMM (NT): Sc[b] = (Q[b] @ K[b]^T) * scale, fp32 out
// ---------------------------------------------------------------------------
__global__ __launch_bounds__(128) void scores_kernel(
    const bf16_t* __restrict__ Q, const bf16_t* __restrict__ Kp,
    float* __restrict__ Sc, float scale) {
  const int b      = blockIdx.z;
  const bf16_t* Ab = Q + (size_t)b * NS * NE;
  const bf16_t* Bb = Kp + (size_t)b * NS * NE;
  float* Cb        = Sc + (size_t)b * NS * NS;

  const int tid  = threadIdx.x;
  const int lane = tid & 31;
  const int wid  = tid >> 5;
  const int wm   = (wid >> 1) * 64;
  const int wn   = (wid & 1) * 64;
  const int gm0  = blockIdx.x * 128;
  const int gn0  = blockIdx.y * 128;

  v8f acc[4][4] = {};
  gemm_core(Ab, NE, Bb, NE, NE, gm0, gn0, tid, lane, wm, wn, acc);

  const int col   = lane & 15;
  const int rbase = (lane < 16) ? 0 : 8;
#pragma unroll
  for (int mf = 0; mf < 4; ++mf)
#pragma unroll
    for (int nf = 0; nf < 4; ++nf) {
      const int gc = gn0 + wn + nf * 16 + col;
#pragma unroll
      for (int r = 0; r < 8; ++r) {
        const int gr = gm0 + wm + mf * 16 + rbase + r;
        Cb[(size_t)gr * NS + gc] = acc[mf][nf][r] * scale;
      }
    }
}

// ---------------------------------------------------------------------------
// Row softmax over Sk: fp32 scores -> bf16 probabilities. One block per row.
// ---------------------------------------------------------------------------
__global__ __launch_bounds__(256) void softmax_kernel(
    const float* __restrict__ Sc, bf16_t* __restrict__ P) {
  __shared__ float red[8];
  const size_t row = blockIdx.x;
  const float* src = Sc + row * NS;
  bf16_t* dst      = P + row * NS;
  const int tid    = threadIdx.x;

  float m = -3.402823466e38f;
  for (int c = tid; c < NS; c += 256) m = fmaxf(m, src[c]);
#pragma unroll
  for (int off = 16; off >= 1; off >>= 1) m = fmaxf(m, __shfl_xor(m, off, 32));
  if ((tid & 31) == 0) red[tid >> 5] = m;
  __syncthreads();
  m = red[0];
#pragma unroll
  for (int i = 1; i < 8; ++i) m = fmaxf(m, red[i]);
  __syncthreads();

  float s = 0.f;
  for (int c = tid; c < NS; c += 256) s += __expf(src[c] - m);
#pragma unroll
  for (int off = 16; off >= 1; off >>= 1) s += __shfl_xor(s, off, 32);
  if ((tid & 31) == 0) red[tid >> 5] = s;
  __syncthreads();
  s = red[0];
#pragma unroll
  for (int i = 1; i < 8; ++i) s += red[i];

  const float inv = 1.0f / s;
  for (int c = tid; c < NS; c += 256)
    dst[c] = (bf16_t)(__expf(src[c] - m) * inv);
}

// ---------------------------------------------------------------------------
// Output GEMM: Out[b] = P[b] (bf16 [S,S]) @ Vt[b]^T (Vt bf16 [E,S]) -> fp32
// ---------------------------------------------------------------------------
__global__ __launch_bounds__(128) void av_kernel(const bf16_t* __restrict__ P,
                                                 const bf16_t* __restrict__ Vt,
                                                 float* __restrict__ Out) {
  const int b      = blockIdx.z;
  const bf16_t* Ab = P + (size_t)b * NS * NS;
  const bf16_t* Bb = Vt + (size_t)b * NE * NS;
  float* Cb        = Out + (size_t)b * NS * NE;

  const int tid  = threadIdx.x;
  const int lane = tid & 31;
  const int wid  = tid >> 5;
  const int wm   = (wid >> 1) * 64;
  const int wn   = (wid & 1) * 64;
  const int gm0  = blockIdx.x * 128;
  const int gn0  = blockIdx.y * 128;

  v8f acc[4][4] = {};
  gemm_core(Ab, NS, Bb, NS, NS, gm0, gn0, tid, lane, wm, wn, acc);

  const int col   = lane & 15;
  const int rbase = (lane < 16) ? 0 : 8;
#pragma unroll
  for (int mf = 0; mf < 4; ++mf)
#pragma unroll
    for (int nf = 0; nf < 4; ++nf) {
      const int gc = gn0 + wn + nf * 16 + col;
#pragma unroll
      for (int r = 0; r < 8; ++r) {
        const int gr = gm0 + wm + mf * 16 + rbase + r;
        Cb[(size_t)gr * NE + gc] = acc[mf][nf][r];
      }
    }
}

// ---------------------------------------------------------------------------
extern "C" void kernel_launch(void* const* d_in, const int* in_sizes, int n_in,
                              void* d_out, int out_size, void* d_ws,
                              size_t ws_size, hipStream_t stream) {
  (void)in_sizes; (void)n_in; (void)out_size; (void)ws_size;

  const float* query = (const float*)d_in[0];
  const float* key   = (const float*)d_in[1];
  const float* value = (const float*)d_in[2];
  const float* Wq    = (const float*)d_in[3];
  const float* bq    = (const float*)d_in[4];
  const float* Wk    = (const float*)d_in[5];
  const float* bk    = (const float*)d_in[6];
  const float* Wv    = (const float*)d_in[7];
  const float* bv    = (const float*)d_in[8];
  float* out         = (float*)d_out;

  const size_t nBSE = (size_t)NB * NS * NE;  // 8388608
  const size_t nEE  = (size_t)NE * NE;       // 1048576
  const size_t nBSS = (size_t)NB * NS * NS;  // 16777216

  char* w    = (char*)d_ws;
  auto alloc = [&](size_t bytes) -> char* {
    char* p = w;
    w += (bytes + 255) & ~(size_t)255;
    return p;
  };
  bf16_t* Xq  = (bf16_t*)alloc(nBSE * 2);
  bf16_t* Xk  = (bf16_t*)alloc(nBSE * 2);
  bf16_t* Xv  = (bf16_t*)alloc(nBSE * 2);
  bf16_t* wqT = (bf16_t*)alloc(nEE * 2);
  bf16_t* wkT = (bf16_t*)alloc(nEE * 2);
  bf16_t* wvT = (bf16_t*)alloc(nEE * 2);
  bf16_t* Qp  = (bf16_t*)alloc(nBSE * 2);
  bf16_t* Kp  = (bf16_t*)alloc(nBSE * 2);
  bf16_t* VtT = (bf16_t*)alloc(nBSE * 2);  // [B][E][S]
  float* Sc   = (float*)alloc(nBSS * 4);
  bf16_t* Pm  = (bf16_t*)alloc(nBSS * 2);

  // Convert activations to bf16 (8 elems/thread, vectorized).
  const int n8x = (int)(nBSE / 8);
  cvt8_kernel<<<(n8x + 255) / 256, 256, 0, stream>>>(query, Xq, n8x);
  cvt8_kernel<<<(n8x + 255) / 256, 256, 0, stream>>>(key, Xk, n8x);
  cvt8_kernel<<<(n8x + 255) / 256, 256, 0, stream>>>(value, Xv, n8x);

  // Convert + transpose weights: Wt[n][k] = (bf16)W[k][n].
  dim3 gt(NE / 32, NE / 32);
  wtcvt_kernel<<<gt, 256, 0, stream>>>(Wq, wqT, NE, NE);
  wtcvt_kernel<<<gt, 256, 0, stream>>>(Wk, wkT, NE, NE);
  wtcvt_kernel<<<gt, 256, 0, stream>>>(Wv, wvT, NE, NE);

  // Projections. V is written pre-transposed as Vt[b][e][s].
  dim3 gp(NB * NS / 128, NE / 128);  // (64, 8)
  proj_kernel<false><<<gp, 128, 0, stream>>>(Xq, wqT, bq, Qp);
  proj_kernel<false><<<gp, 128, 0, stream>>>(Xk, wkT, bk, Kp);
  proj_kernel<true><<<gp, 128, 0, stream>>>(Xv, wvT, bv, VtT);

  // Scores: Q @ K^T / sqrt(E).
  dim3 gs(NS / 128, NS / 128, NB);  // (16, 16, 4)
  scores_kernel<<<gs, 128, 0, stream>>>(Qp, Kp, Sc, 0.03125f);

  // Softmax rows (B*S rows of length S).
  softmax_kernel<<<NB * NS, 256, 0, stream>>>(Sc, Pm);

  // Out = P @ V   (V supplied as Vt[b][e][s], K-contiguous).
  dim3 ga(NS / 128, NE / 128, NB);  // (16, 8, 4)
  av_kernel<<<ga, 128, 0, stream>>>(Pm, VtT, out);
}
